// SPDConv2d_41059887350426
// MI455X (gfx1250) — matrix operand order repaired and checked by reading, not compiled, not run on hardware
//
#include <hip/hip_runtime.h>
#include <hip/hip_bf16.h>

typedef __attribute__((ext_vector_type(16))) __bf16        v16bf;
typedef __attribute__((ext_vector_type(8)))  float         v8f;
typedef __attribute__((ext_vector_type(2)))  float         v2f;
typedef __attribute__((ext_vector_type(4)))  unsigned int  v4u;

#define EPS_W     1e-6f
#define CIN       16
#define COUT      16
#define BATCH     64
#define N_IN      256
#define M_OUT     254
#define TILE_ROWS 8
#define IN_ROWS   (TILE_ROWS + 2)
#define NCHUNK    5            // ceil(9 taps / 2 taps-per-K32-chunk)

// Workspace layout (float units):
//   ws[0..15]            : per-output-channel sum of squares (zeroed by prep each call)
//   byte offset 256      : B fragments, hi plane: 5 chunks * 32 lanes * 32B = 5120 B
//   byte offset 256+5120 : B fragments, lo plane: 5120 B
// Per-lane fragment layout (matches 16-bit WMMA operand striping):
//   lane n<16 -> column n, K=0..7,16..23 packed as 8 dwords; lane n>=16 -> column n-16, K=8..15,24..31.

union Frag32B { v4u u[2]; v16bf v; };

// ---------------------------------------------------------------------------
// Prep: W[o,i,b,c] = sum_a V[o,i,a,b]*V[o,i,a,c] + eps*(b==c), split into
// bf16 hi/lo and scattered directly into the per-lane WMMA B-fragment layout.
// Also zero the 16 channel sum-of-squares accumulators.
// ---------------------------------------------------------------------------
__global__ void spd_prep(const float* __restrict__ V, float* __restrict__ ws) {
  const int tid = threadIdx.x;
  if (tid < 16) ws[tid] = 0.0f;
  __bf16* pb = (__bf16*)(ws + 64);              // halfword-indexed fragment region
  for (int idx = tid; idx < NCHUNK * 512; idx += 256) {
    const int chunk = idx >> 9;
    const int rem   = idx & 511;
    const int lane  = rem >> 4;
    const int h     = rem & 15;
    const int n     = lane & 15;                // B column = output channel
    const int K     = ((h >> 3) << 4) | ((lane >> 4) << 3) | (h & 7);   // K within chunk: 0..31
    const int t     = K >> 4;                   // tap within chunk (0/1)
    const int ci    = K & 15;                   // input channel
    const int tap   = chunk * 2 + t;
    float w = 0.0f;
    if (tap < 9) {
      const int ty = tap / 3, tx = tap % 3;
      const float* vb = V + (n * CIN + ci) * 9;
      #pragma unroll
      for (int a = 0; a < 3; ++a) w += vb[a * 3 + ty] * vb[a * 3 + tx];
      if (ty == tx) w += EPS_W;
    }
    const __bf16 hi = (__bf16)w;
    const __bf16 lo = (__bf16)(w - (float)hi);
    pb[idx]                = hi;
    pb[NCHUNK * 512 + idx] = lo;
  }
}

// ---------------------------------------------------------------------------
// Main conv: block = 256 threads (8 waves), handles (batch b, 8 output rows).
// LDS tile: 10 input rows x 256 cols x 16ch, bf16 hi+lo planes, pixel-major
// (64B per pixel) so each A fragment half is one ds_load_b128.
// Each wave: 1 output row, 16 x-tiles x (5 chunks x 3 wmma) bf16 WMMAs.
// D tiles are transposed through a per-wave LDS patch so global stores are
// full-64B-line coalesced (row stride 254 floats => only 8B alignment, so b64).
// ---------------------------------------------------------------------------
__global__ __launch_bounds__(256) void spd_conv(const float* __restrict__ X,
                                                float* __restrict__ ws,
                                                float* __restrict__ out) {
  __shared__ v4u   lds[IN_ROWS * 256 * 4];      // 160 KB input tile
  __shared__ float stage[8][16][17];            // per-wave D transpose patch
  __shared__ float sh_ssq[16];
  const int tid = threadIdx.x;
  const int b   = blockIdx.x;
  const int y0  = blockIdx.y * TILE_ROWS;
  if (tid < 16) sh_ssq[tid] = 0.0f;

  // ---- stage input tile: fp32 -> (bf16 hi, bf16 lo) ----
  {
    __bf16* l16 = (__bf16*)lds;
    const int c = tid;                          // 256 threads == 256 columns
    for (int ch = 0; ch < CIN; ++ch) {
      const float* src = X + ((size_t)(ch * BATCH + b) << 16);
      #pragma unroll
      for (int r = 0; r < IN_ROWS; ++r) {
        int gr = y0 + r; gr = gr > (N_IN - 1) ? (N_IN - 1) : gr;
        const float x = src[(gr << 8) | c];
        const __bf16 hi = (__bf16)x;
        const __bf16 lo = (__bf16)(x - (float)hi);
        const int p = ((r << 8) | c) << 5;      // halfword base of pixel
        l16[p + ch]      = hi;
        l16[p + 16 + ch] = lo;
      }
    }
  }

  // ---- per-lane resident B fragments (hi + lo) ----
  const int lane = tid & 31;
  const int wave = tid >> 5;
  v16bf bh[NCHUNK], bl[NCHUNK];
  {
    const v4u* bsrc = (const v4u*)(ws + 64);
    #pragma unroll
    for (int q = 0; q < NCHUNK; ++q) {
      Frag32B fh, fl;
      fh.u[0] = bsrc[(q * 32 + lane) * 2 + 0];
      fh.u[1] = bsrc[(q * 32 + lane) * 2 + 1];
      fl.u[0] = bsrc[320 + (q * 32 + lane) * 2 + 0];
      fl.u[1] = bsrc[320 + (q * 32 + lane) * 2 + 1];
      bh[q] = fh.v;
      bl[q] = fl.v;
    }
  }
  __syncthreads();

  const int row  = y0 + wave;                   // this wave's output row
  const int ln15 = lane & 15;                   // A-row M and D-column N
  const int hs   = lane >> 4;                   // K-half select (+16B in LDS)
  const int csub = lane >> 3;                   // 0..3: channel within store group
  const int xp   = lane & 7;                    // x pair within store group
  float ssq = 0.0f;

  for (int xt = 0; xt < 16; ++xt) {
    const int x0 = xt << 4;
    v8f acc = {0.f, 0.f, 0.f, 0.f, 0.f, 0.f, 0.f, 0.f};
    #pragma unroll
    for (int q = 0; q < NCHUNK; ++q) {
      const int t0 = q * 2, t1 = q * 2 + 1;
      const int ty0 = t0 / 3, tx0 = t0 % 3;
      int c0 = x0 + ln15 + tx0; c0 = c0 > (N_IN - 1) ? (N_IN - 1) : c0;
      const int p0 = (((wave + ty0) << 8) + c0) << 2;   // v4u index of pixel
      Frag32B ah, al;
      ah.u[0] = lds[p0 + hs];
      al.u[0] = lds[p0 + 2 + hs];
      if (t1 < 9) {
        const int ty1 = t1 / 3, tx1 = t1 % 3;
        int c1 = x0 + ln15 + tx1; c1 = c1 > (N_IN - 1) ? (N_IN - 1) : c1;
        const int p1 = (((wave + ty1) << 8) + c1) << 2;
        ah.u[1] = lds[p1 + hs];
        al.u[1] = lds[p1 + 2 + hs];
      } else {
        const v4u z = {0u, 0u, 0u, 0u};
        ah.u[1] = z;
        al.u[1] = z;
      }
      // D += A_hi*B_hi + A_hi*B_lo + A_lo*B_hi   (error-compensated bf16)
      acc = __builtin_amdgcn_wmma_f32_16x16x32_bf16(false, ah.v, false, bh[q],
                                                    (short)0, acc, false, false);
      acc = __builtin_amdgcn_wmma_f32_16x16x32_bf16(false, ah.v, false, bl[q],
                                                    (short)0, acc, false, false);
      acc = __builtin_amdgcn_wmma_f32_16x16x32_bf16(false, al.v, false, bh[q],
                                                    (short)0, acc, false, false);
    }
    // D layout: VGPR r, lane<16 -> (M=r, N=lane); lane>=16 -> (M=r+8, N=lane-16)
    if (row < M_OUT) {
      if (xt < 15) {
        // Full tile: transpose via per-wave LDS patch, then coalesced b64
        // stores (8 consecutive lanes cover one full 64B line of one channel).
        #pragma unroll
        for (int r = 0; r < 8; ++r) {
          const int M = (hs << 3) + r;
          stage[wave][M][ln15] = acc[r];
          ssq += acc[r] * acc[r];
        }
        __builtin_amdgcn_wave_barrier();
        #pragma unroll
        for (int cg = 0; cg < 4; ++cg) {
          const int c = cg * 4 + csub;
          const v2f v = { stage[wave][xp * 2 + 0][c],
                          stage[wave][xp * 2 + 1][c] };
          float* pdst = out + ((b * COUT + c) * M_OUT + row) * M_OUT
                            + x0 + xp * 2;
          *(v2f*)pdst = v;
        }
        __builtin_amdgcn_wave_barrier();
      } else {
        // Partial tile (x = 240..253): masked scalar stores.
        float* obase = out + ((b * COUT + ln15) * M_OUT + row) * M_OUT;
        #pragma unroll
        for (int r = 0; r < 8; ++r) {
          const int x = x0 + (hs << 3) + r;
          if (x < M_OUT) {
            const float v = acc[r];
            obase[x] = v;
            ssq += v * v;
          }
        }
      }
    }
  }

  atomicAdd(&sh_ssq[ln15], ssq);                // ds_add_f32 block reduce
  __syncthreads();
  if (tid < 16) atomicAdd(&ws[tid], sh_ssq[tid]);
}

// ---------------------------------------------------------------------------
// Finalize: out = relu(y * rsqrt(ssq[ch])), in place.
// grid.y = b*16+ch (1024), grid.x covers the 254*254 plane.
// ---------------------------------------------------------------------------
__global__ void spd_scale(float* __restrict__ out, const float* __restrict__ ws) {
  const int plane = M_OUT * M_OUT;              // 64516
  const int i = blockIdx.x * 256 + threadIdx.x;
  if (i >= plane) return;
  const int ch = blockIdx.y & 15;
  const float s = 1.0f / sqrtf(ws[ch]);
  const size_t idx = (size_t)blockIdx.y * plane + i;
  const float v = out[idx] * s;
  out[idx] = v > 0.0f ? v : 0.0f;
}

extern "C" void kernel_launch(void* const* d_in, const int* in_sizes, int n_in,
                              void* d_out, int out_size, void* d_ws, size_t ws_size,
                              hipStream_t stream) {
  const float* X = (const float*)d_in[0];   // [16, 64, 256, 256] fp32
  const float* V = (const float*)d_in[1];   // [16, 16, 3, 3] fp32
  float* out = (float*)d_out;               // [64, 16, 254, 254] fp32
  float* ws  = (float*)d_ws;

  spd_prep<<<1, 256, 0, stream>>>(V, ws);
  spd_conv<<<dim3(BATCH, N_IN / TILE_ROWS), 256, 0, stream>>>(X, ws, out);
  const int plane_blocks = (M_OUT * M_OUT + 255) / 256;   // 253
  spd_scale<<<dim3(plane_blocks, BATCH * COUT), 256, 0, stream>>>(out, ws);
}
